// ManualRPCALinear_49787260895356
// MI455X (gfx1250) — matrix operand order repaired and checked by reading, not compiled, not run on hardware
//
#include <hip/hip_runtime.h>
#include <hip/hip_bf16.h>

// MI455X / gfx1250, wave32. Main GEMMs run on v_wmma_f32_16x16x32_bf16.
// Problem: B=4,S=2048 -> M=8192 rows; D_IN=K=4096; D_OUT=N=4096; RANK=32.
// Compute-bound (~0.55 TFLOP vs ~17us of HBM traffic at 23.3 TB/s), so the
// design goal is maximum WMMA issue density: 64x64 tile per wave (4x4 WMMA
// accumulators) -> 16 WMMAs per 8 fragment loads per K-step per path.
// Workspace requirement: ~257 MB (bf16 staged activations + weights + LoRA).

#define M_DIM 8192
#define N_DIM 4096
#define K_DIM 4096
#define RANK  32

typedef __attribute__((ext_vector_type(16))) __bf16 v16bf;
typedef __attribute__((ext_vector_type(8)))  __bf16 v8bf;
typedef __attribute__((ext_vector_type(8)))  float  v8f;

// ---------------------------------------------------------------------------
// Fragment loaders (wave32, 16-bit operands); this code's contiguous K order, same for A and B.
// A-matrix 16x32 (MxK): lanes 0-15 row M=lane hold K 0..15 (packed 2/VGPR);
//                       lanes 16-31 row M=lane-16 hold K 16..31.
// B-matrix 32x16 (KxN): lanes 0-15 col N=lane hold K 0..15 (packed 2/VGPR);
//                       lanes 16-31 col N=lane-16 hold K 16..31.
// Both activations (row-major [M,K]) and weights (row-major [N,K]) are
// K-contiguous, so every lane issues 16-byte global loads.
// ---------------------------------------------------------------------------
__device__ __forceinline__ v16bf load_a_frag(const __bf16* __restrict__ A, int ld,
                                             int row0, int k, int lane) {
  const int r  = row0 + (lane & 15);
  const int kb = k + ((lane >> 4) << 4);      // +0 or +16
  const __bf16* p = A + (size_t)r * ld + kb;
  v8bf lo = *(const v8bf*)p;                  // K kb..kb+7
  v8bf hi = *(const v8bf*)(p + 8);            // K kb+8..kb+15
  return __builtin_shufflevector(lo, hi, 0,1,2,3,4,5,6,7,8,9,10,11,12,13,14,15);
}

__device__ __forceinline__ v16bf load_b_frag(const __bf16* __restrict__ W, int ld,
                                             int col0, int k, int lane) {
  const int c  = col0 + (lane & 15);
  const int kb = k + ((lane >> 4) << 4);      // +0 or +16
  const __bf16* p = W + (size_t)c * ld + kb;
  v8bf lo = *(const v8bf*)p;                  // K kb..kb+7
  v8bf hi = *(const v8bf*)(p + 8);            // K kb+8..kb+15
  return __builtin_shufflevector(lo, hi, 0,1,2,3,4,5,6,7,8,9,10,11,12,13,14,15);
}

__device__ __forceinline__ v8f wmma_bf16(v16bf a, v16bf b, v8f c) {
  return __builtin_amdgcn_wmma_f32_16x16x32_bf16(
      /*neg_a=*/false, a, /*neg_b=*/false, b,
      /*c_mod=*/(short)0, c, /*reuse_a=*/false, /*reuse_b=*/false);
}

// ---------------------------------------------------------------------------
// Kernel 1: smoothing + block-16 symmetric INT4 fake-quant, emit bf16 stages.
// 16-lane groups (aligned with quant blocks) reduce |x_s| via shfl_xor.
// ---------------------------------------------------------------------------
__global__ __launch_bounds__(256)
void prep_act_kernel(const float* __restrict__ x, const float* __restrict__ ss,
                     __bf16* __restrict__ xq, __bf16* __restrict__ xs,
                     __bf16* __restrict__ xb) {
  const long long i = (long long)blockIdx.x * blockDim.x + threadIdx.x;
  const int kcol = (int)(i & (K_DIM - 1));
  const float xv = x[i];
  const float sm = xv * ss[kcol];
  float a = fabsf(sm);
  a = fmaxf(a, __shfl_xor(a, 1));
  a = fmaxf(a, __shfl_xor(a, 2));
  a = fmaxf(a, __shfl_xor(a, 4));
  a = fmaxf(a, __shfl_xor(a, 8));   // amax over the 16-element quant block
  a = fmaxf(a, 1e-12f);
  const float scale = a * (1.0f / 7.0f);
  float q = rintf(sm / scale);
  q = fminf(fmaxf(q, -7.0f), 7.0f) * scale;
  xq[i] = (__bf16)q;
  xs[i] = (__bf16)sm;
  xb[i] = (__bf16)xv;
}

// ---------------------------------------------------------------------------
// Kernel 2: fp32 -> bf16 weight staging.
// ---------------------------------------------------------------------------
__global__ __launch_bounds__(256)
void cvt_bf16_kernel(const float* __restrict__ in, __bf16* __restrict__ out,
                     long long n) {
  const long long i = (long long)blockIdx.x * blockDim.x + threadIdx.x;
  if (i < n) out[i] = (__bf16)in[i];
}

// ---------------------------------------------------------------------------
// Kernel 3: As = x_smoothed @ lora_a^T  ([M,K]x[RANK,K] -> [M,RANK], bf16 out).
// One wave per 16-row tile, N = 32 = two 16-wide tiles.
// ---------------------------------------------------------------------------
__global__ __launch_bounds__(32)
void lora_pass1_kernel(const __bf16* __restrict__ Xs, const __bf16* __restrict__ La,
                       __bf16* __restrict__ As) {
  const int lane = threadIdx.x;
  const int m0 = blockIdx.x * 16;
  v8f acc0 = {}, acc1 = {};
  for (int k = 0; k < K_DIM; k += 32) {
    v16bf a  = load_a_frag(Xs, K_DIM, m0, k, lane);
    v16bf b0 = load_b_frag(La, K_DIM, 0,  k, lane);
    v16bf b1 = load_b_frag(La, K_DIM, 16, k, lane);
    acc0 = wmma_bf16(a, b0, acc0);
    acc1 = wmma_bf16(a, b1, acc1);
  }
  // C layout: lanes 0-15 -> N=lane, rows m0+0..7; lanes 16-31 -> rows m0+8..15.
  const int col   = lane & 15;
  const int rbase = m0 + ((lane >> 4) << 3);
  #pragma unroll
  for (int r = 0; r < 8; ++r) {
    As[(size_t)(rbase + r) * RANK + col]      = (__bf16)acc0[r];
    As[(size_t)(rbase + r) * RANK + 16 + col] = (__bf16)acc1[r];
  }
}

// ---------------------------------------------------------------------------
// Kernel 4: fused main GEMM.
//   out = xq@Wq^T + x@Ws^T + As@Lb^T + bias
// One wave per 64x64 output tile: 4x4 WMMA accumulators (128 VGPRs) so each
// K-step does 16 WMMAs per 8 fragment loads per path (1 b128 load : 1 WMMA),
// keeping the XDL pipe fed. Wq+Ws bf16 (64 MB) fit the 192 MB L2, so weight
// re-reads across the 64 N-blocks stay on-chip.
// ---------------------------------------------------------------------------
__global__ __launch_bounds__(32)
void rpca_gemm_kernel(const __bf16* __restrict__ Aq, const __bf16* __restrict__ Ax,
                      const __bf16* __restrict__ Wq, const __bf16* __restrict__ Wsp,
                      const __bf16* __restrict__ As, const __bf16* __restrict__ Lb,
                      const float* __restrict__ bias, float* __restrict__ out) {
  const int lane = threadIdx.x;
  const int n0 = blockIdx.x * 64;
  const int m0 = blockIdx.y * 64;
  v8f acc[4][4] = {};

  for (int k = 0; k < K_DIM; k += 32) {
    // prefetch next K-slab of the quantized-path operands (global_prefetch_b8)
    if (k + 64 < K_DIM) {
      __builtin_prefetch(Aq + (size_t)(m0 + (lane & 15)) * K_DIM + k + 64, 0, 1);
      __builtin_prefetch(Wq + (size_t)(n0 + (lane & 15)) * K_DIM + k + 64, 0, 1);
    }
    v16bf af[4], bw[4];
    // quantized main path
    #pragma unroll
    for (int i = 0; i < 4; ++i) af[i] = load_a_frag(Aq, K_DIM, m0 + 16 * i, k, lane);
    #pragma unroll
    for (int j = 0; j < 4; ++j) bw[j] = load_b_frag(Wq, K_DIM, n0 + 16 * j, k, lane);
    #pragma unroll
    for (int i = 0; i < 4; ++i)
      #pragma unroll
      for (int j = 0; j < 4; ++j)
        acc[i][j] = wmma_bf16(af[i], bw[j], acc[i][j]);
    // sparse-outlier path (unsmoothed x)
    #pragma unroll
    for (int i = 0; i < 4; ++i) af[i] = load_a_frag(Ax, K_DIM, m0 + 16 * i, k, lane);
    #pragma unroll
    for (int j = 0; j < 4; ++j) bw[j] = load_b_frag(Wsp, K_DIM, n0 + 16 * j, k, lane);
    #pragma unroll
    for (int i = 0; i < 4; ++i)
      #pragma unroll
      for (int j = 0; j < 4; ++j)
        acc[i][j] = wmma_bf16(af[i], bw[j], acc[i][j]);
  }

  // LoRA epilogue: single K=32 WMMA per tile pair (rank-32 folded in here so
  // we never pay a full 4096x4096 GEMM for a rank-32 correction).
  {
    v16bf la[4], lb[4];
    #pragma unroll
    for (int i = 0; i < 4; ++i) la[i] = load_a_frag(As, RANK, m0 + 16 * i, 0, lane);
    #pragma unroll
    for (int j = 0; j < 4; ++j) lb[j] = load_b_frag(Lb, RANK, n0 + 16 * j, 0, lane);
    #pragma unroll
    for (int i = 0; i < 4; ++i)
      #pragma unroll
      for (int j = 0; j < 4; ++j)
        acc[i][j] = wmma_bf16(la[i], lb[j], acc[i][j]);
  }

  // bias + store. C layout: lanes 0-15 -> N=lane, M rows 0..7; lanes 16-31 -> M 8..15.
  const int colOff = lane & 15;
  const int rOff   = (lane >> 4) << 3;
  #pragma unroll
  for (int j = 0; j < 4; ++j) {
    const int col = n0 + j * 16 + colOff;
    const float bv = bias[col];
    #pragma unroll
    for (int i = 0; i < 4; ++i) {
      const int rbase = m0 + i * 16 + rOff;
      #pragma unroll
      for (int r = 0; r < 8; ++r)
        out[(size_t)(rbase + r) * N_DIM + col] = acc[i][j][r] + bv;
    }
  }
}

// ---------------------------------------------------------------------------
// Host launcher. Inputs (setup_inputs order):
//   0:x [M,K] f32   1:smooth_scale [K] f32   2:w_quantized [N,K] f32
//   3:lora_a [32,K] f32   4:lora_b [N,32] f32   5:w_sparse [N,K] f32
//   6:bias [N] f32       out: [M,N] f32
// ---------------------------------------------------------------------------
extern "C" void kernel_launch(void* const* d_in, const int* in_sizes, int n_in,
                              void* d_out, int out_size, void* d_ws, size_t ws_size,
                              hipStream_t stream) {
  (void)in_sizes; (void)n_in; (void)out_size; (void)ws_size;

  const float* x    = (const float*)d_in[0];
  const float* ss   = (const float*)d_in[1];
  const float* wq_f = (const float*)d_in[2];
  const float* la_f = (const float*)d_in[3];
  const float* lb_f = (const float*)d_in[4];
  const float* wsp_f= (const float*)d_in[5];
  const float* bias = (const float*)d_in[6];
  float* out = (float*)d_out;

  // Workspace carve-up (bytes, 256-aligned).
  char* ws = (char*)d_ws;
  const size_t act_elems = (size_t)M_DIM * K_DIM;      // 33,554,432
  const size_t w_elems   = (size_t)N_DIM * K_DIM;
  __bf16* xq  = (__bf16*)(ws);                                  // 64 MB
  __bf16* xs  = (__bf16*)(ws + act_elems * 2);                  // 64 MB
  __bf16* xb  = (__bf16*)(ws + act_elems * 4);                  // 64 MB
  __bf16* wqb = (__bf16*)(ws + act_elems * 6);                  // 32 MB
  __bf16* wspb= (__bf16*)(ws + act_elems * 6 + w_elems * 2);    // 32 MB
  char* tail  = ws + act_elems * 6 + w_elems * 4;
  __bf16* lab = (__bf16*)(tail);                                // 256 KB
  __bf16* lbb = (__bf16*)(tail + (size_t)RANK * K_DIM * 2);     // 256 KB
  __bf16* asb = (__bf16*)(tail + (size_t)RANK * K_DIM * 2
                               + (size_t)N_DIM * RANK * 2);     // 512 KB

  // 1) smoothing + fake-quant + bf16 staging of activations
  prep_act_kernel<<<dim3((unsigned)(act_elems / 256)), dim3(256), 0, stream>>>(
      x, ss, xq, xs, xb);

  // 2) bf16 staging of weights / LoRA factors
  cvt_bf16_kernel<<<dim3((unsigned)((w_elems + 255) / 256)), dim3(256), 0, stream>>>(
      wq_f, wqb, (long long)w_elems);
  cvt_bf16_kernel<<<dim3((unsigned)((w_elems + 255) / 256)), dim3(256), 0, stream>>>(
      wsp_f, wspb, (long long)w_elems);
  cvt_bf16_kernel<<<dim3((unsigned)(((size_t)RANK * K_DIM + 255) / 256)), dim3(256), 0, stream>>>(
      la_f, lab, (long long)RANK * K_DIM);
  cvt_bf16_kernel<<<dim3((unsigned)(((size_t)N_DIM * RANK + 255) / 256)), dim3(256), 0, stream>>>(
      lb_f, lbb, (long long)N_DIM * RANK);

  // 3) LoRA first contraction: As = x_smoothed @ lora_a^T
  lora_pass1_kernel<<<dim3(M_DIM / 16), dim3(32), 0, stream>>>(xs, lab, asb);

  // 4) fused main GEMM + LoRA epilogue + bias (64x64 tile per wave)
  rpca_gemm_kernel<<<dim3(N_DIM / 64, M_DIM / 64), dim3(32), 0, stream>>>(
      xq, xb, wqb, wspb, asb, lbb, bias, out);
}
